// SwinTransformerBlock3D_53291954209308
// MI455X (gfx1250) — matrix operand order repaired and checked by reading, not compiled, not run on hardware
//
#include <hip/hip_runtime.h>
#include <hip/hip_bf16.h>
#include <math.h>

// ---------------- problem constants ----------------
#define BD   16          // D
#define BH   56          // H
#define BW   56          // W
#define CC   128         // C
#define NHD  4           // heads
#define HEAD 32
#define NTOK 392         // tokens per window (8*7*7)
#define NP   416         // padded tokens per window (26*16 = 13*32)
#define NWIN 128         // number of windows (2*8*8)
#define HID  512
#define TOK  50176       // B*D*H*W tokens
#define MPAD 53248       // NWIN*NP rows
#define SCALE_Q 0.17677669529663687f   // 32^-0.5
#define NEG_BIG (-1.0e9f)

typedef _Float16 half_t;
typedef __attribute__((ext_vector_type(16))) _Float16 v16h;
typedef __attribute__((ext_vector_type(8)))  float    v8f;

// load 8 contiguous halves (16B) into lanes of a v16h fragment
template <int OFF>
__device__ inline void load8h(v16h& dst, const half_t* p) {
  union { uint4 u; _Float16 h[8]; } cv;
  cv.u = *reinterpret_cast<const uint4*>(p);
#pragma unroll
  for (int i = 0; i < 8; ++i) dst[OFF + i] = cv.h[i];
}

__device__ inline v8f wmma_f16(const v16h& a, const v16h& b, const v8f& c) {
  return __builtin_amdgcn_wmma_f32_16x16x32_f16(false, a, false, b, (short)0, c,
                                                false, false);
}

// ---------------- setup kernels ----------------
__global__ void cvt_f32_f16(const float* __restrict__ s, half_t* __restrict__ d, int n) {
  int i = blockIdx.x * blockDim.x + threadIdx.x;
  if (i < n) d[i] = (half_t)s[i];
}

// bias[h][m][n] = rel_pos_table[REL_IDX[m][n]][h]  (NEG_BIG on padded key cols/rows)
__global__ void build_bias(const float* __restrict__ relTab, float* __restrict__ biasT) {
  int i = blockIdx.x * blockDim.x + threadIdx.x;
  int tot = NHD * NP * NP;
  if (i >= tot) return;
  int h = i / (NP * NP);
  int r = i % (NP * NP);
  int m = r / NP, n = r % NP;
  float b;
  if (m >= NTOK || n >= NTOK) {
    b = NEG_BIG;
  } else {
    int td = m / 49, th = (m % 49) / 7, tw = m % 7;
    int sd = n / 49, sh = (n % 49) / 7, sw = n % 7;
    int dt = td - sd + 7, dh = th - sh + 6, dw = tw - sw + 6;
    int ridx = (dt * 13 + dh) * 13 + dw;
    b = relTab[ridx * NHD + h];
  }
  biasT[i] = b;
}

// region id per (window, token) for the shifted-window mask
__global__ void build_regid(int* __restrict__ regid) {
  int i = blockIdx.x * blockDim.x + threadIdx.x;
  if (i >= NWIN * NP) return;
  int w = i / NP, n = i % NP;
  int out = 0;
  if (n < NTOK) {
    int td = n / 49, rr = n % 49, th = rr / 7, tw = rr % 7;
    int wd = w >> 6, wh = (w >> 3) & 7, ww = w & 7;
    int d = wd * 8 + td, hh = wh * 7 + th, wp = ww * 7 + tw;
    int rd = d < 8 ? 0 : (d < 12 ? 1 : 2);
    int rh = hh < 49 ? 0 : (hh < 53 ? 1 : 2);
    int rw = wp < 49 ? 0 : (wp < 53 ? 1 : 2);
    out = rd * 9 + rh * 3 + rw;
  }
  regid[i] = out;
}

// ---------------- LayerNorm1 + cyclic shift + window partition -> f16 ----------------
__global__ __launch_bounds__(128) void ln1_shift_win(const float* __restrict__ x,
                                                     const float* __restrict__ gw,
                                                     const float* __restrict__ gb,
                                                     half_t* __restrict__ xw) {
  int wn = blockIdx.x;            // 0 .. MPAD-1
  int w = wn / NP, n = wn % NP;
  int c = threadIdx.x;
  half_t* orow = xw + (size_t)wn * CC;
  if (n >= NTOK) { orow[c] = (half_t)0.0f; return; }
  int td = n / 49, rr = n % 49, th = rr / 7, tw = rr % 7;
  int wd = w >> 6, wh = (w >> 3) & 7, ww = w & 7;
  int d = (wd * 8 + td + 4) & 15;              // roll -SD
  int hh = wh * 7 + th + 3; if (hh >= 56) hh -= 56;
  int wp = ww * 7 + tw + 3; if (wp >= 56) wp -= 56;
  const float* row = x + (((size_t)d * 56 + hh) * 56 + wp) * CC;
  float v = row[c];
  __shared__ float red[128];
  red[c] = v; __syncthreads();
  for (int s = 64; s > 0; s >>= 1) { if (c < s) red[c] += red[c + s]; __syncthreads(); }
  float mu = red[0] * (1.0f / 128.0f);
  __syncthreads();
  float dv = v - mu;
  red[c] = dv * dv; __syncthreads();
  for (int s = 64; s > 0; s >>= 1) { if (c < s) red[c] += red[c + s]; __syncthreads(); }
  float var = red[0] * (1.0f / 128.0f);
  float rs = rsqrtf(var + 1e-5f);
  orow[c] = (half_t)(dv * rs * gw[c] + gb[c]);
}

// ---------------- LayerNorm2 (plain tokens) ----------------
__global__ __launch_bounds__(128) void ln2_plain(const float* __restrict__ xin,
                                                 const float* __restrict__ gw,
                                                 const float* __restrict__ gb,
                                                 half_t* __restrict__ m1) {
  int t = blockIdx.x;             // 0 .. TOK-1
  int c = threadIdx.x;
  float v = xin[(size_t)t * CC + c];
  __shared__ float red[128];
  red[c] = v; __syncthreads();
  for (int s = 64; s > 0; s >>= 1) { if (c < s) red[c] += red[c + s]; __syncthreads(); }
  float mu = red[0] * (1.0f / 128.0f);
  __syncthreads();
  float dv = v - mu;
  red[c] = dv * dv; __syncthreads();
  for (int s = 64; s > 0; s >>= 1) { if (c < s) red[c] += red[c + s]; __syncthreads(); }
  float var = red[0] * (1.0f / 128.0f);
  float rs = rsqrtf(var + 1e-5f);
  m1[(size_t)t * CC + c] = (half_t)(dv * rs * gw[c] + gb[c]);
}

// ---------------- generic NT GEMM: out[m][n] = sum_k A[m][k] * Wt[n][k] ----------------
// block = 256 threads = 8 waves; each wave computes a 32x64 tile (2 M-tiles x 4 N-tiles,
// B fragments shared by both A fragments -> 1.5 b128 loads per WMMA).
// mode 0: QKV split  (oh0=q scaled, oh1=k, oh2=vT transposed)
// mode 1: proj + window-reverse + roll + residual -> outf (=x1 image, fp32)
// mode 2: fc1 + exact GELU -> oh0 (f16)
// mode 3: fc2 + residual add into outf
__global__ __launch_bounds__(256) void gemm_nt(const half_t* __restrict__ A,
                                               const half_t* __restrict__ Wt,
                                               const float* __restrict__ bias,
                                               int K, int mode,
                                               float* __restrict__ outf,
                                               const float* __restrict__ xin,
                                               half_t* __restrict__ oh0,
                                               half_t* __restrict__ oh1,
                                               half_t* __restrict__ oh2) {
  int lane = threadIdx.x & 31, wv = threadIdx.x >> 5;
  int l16 = lane >> 4, lm = lane & 15;
  size_t mBase = (size_t)blockIdx.x * 256 + (size_t)wv * 32;
  int nBase = blockIdx.y * 64;
  v8f acc[2][4] = {};
  const half_t* arow0 = A + (mBase + lm) * (size_t)K;
  const half_t* arow1 = A + (mBase + 16 + lm) * (size_t)K;
  for (int kb = 0; kb < K; kb += 32) {
    v16h af0, af1;
    load8h<0>(af0, arow0 + kb + 16 * l16);
    load8h<8>(af0, arow0 + kb + 16 * l16 + 8);
    load8h<0>(af1, arow1 + kb + 16 * l16);
    load8h<8>(af1, arow1 + kb + 16 * l16 + 8);
#pragma unroll
    for (int t = 0; t < 4; ++t) {
      const half_t* wrow = Wt + (size_t)(nBase + t * 16 + lm) * K + kb + 16 * l16;
      v16h bf;
      load8h<0>(bf, wrow);
      load8h<8>(bf, wrow + 8);
      acc[0][t] = wmma_f16(af0, bf, acc[0][t]);
      acc[1][t] = wmma_f16(af1, bf, acc[1][t]);
    }
  }
#pragma unroll
  for (int mt = 0; mt < 2; ++mt) {
#pragma unroll
    for (int t = 0; t < 4; ++t) {
      int n = nBase + t * 16 + lm;
#pragma unroll
      for (int vj = 0; vj < 8; ++vj) {
        size_t m = mBase + mt * 16 + vj + 8 * l16;
        float val = acc[mt][t][vj];
        if (mode == 0) {                       // QKV
          float bv = bias[n];
          if (n < 128) {
            oh0[m * CC + n] = (half_t)((val + bv) * SCALE_Q);
          } else if (n < 256) {
            oh1[m * CC + (n - 128)] = (half_t)(val + bv);
          } else {
            int cc = n - 256, hh = cc >> 5, ch = cc & 31;
            size_t w = m / NP, tok = m % NP;
            oh2[((w * NHD + hh) * HEAD + ch) * NP + tok] = (half_t)(val + bv);
          }
        } else if (mode == 1) {                // proj + reverse + roll + residual
          int w = (int)(m / NP), tok = (int)(m % NP);
          if (tok < NTOK) {
            int td = tok / 49, rr = tok % 49, th = rr / 7, tw = rr % 7;
            int wd = w >> 6, wh = (w >> 3) & 7, ww = w & 7;
            int d = (wd * 8 + td + 4) & 15;    // roll +SD
            int hh = wh * 7 + th + 3; if (hh >= 56) hh -= 56;
            int wp = ww * 7 + tw + 3; if (wp >= 56) wp -= 56;
            size_t oi = (((size_t)d * 56 + hh) * 56 + wp) * CC + n;
            outf[oi] = xin[oi] + val + bias[n];
          }
        } else if (mode == 2) {                // fc1 + GELU(exact)
          float u = val + bias[n];
          float g = 0.5f * u * (1.0f + erff(u * 0.70710678118f));
          oh0[m * (size_t)HID + n] = (half_t)g;
        } else {                               // fc2 + residual
          size_t oi = m * CC + n;
          outf[oi] = outf[oi] + val + bias[n];
        }
      }
    }
  }
}

// ---------------- flash attention: one wave per (window, head, 16-row q tile) ---------
__global__ __launch_bounds__(32) void attn_flash(const half_t* __restrict__ q,
                                                 const half_t* __restrict__ k,
                                                 const half_t* __restrict__ vT,
                                                 const float* __restrict__ biasT,
                                                 const int* __restrict__ regid,
                                                 half_t* __restrict__ y) {
  int qt = blockIdx.x;   // 0..25
  int h  = blockIdx.y;   // 0..3
  int w  = blockIdx.z;   // 0..127
  int lane = threadIdx.x & 31;
  int l16 = lane >> 4, lm = lane & 15;
  __shared__ _Float16 pSh[16][32];

  // Q fragment (A layout, K-dim = HEAD = 32)
  const half_t* qp = q + ((size_t)w * NP + qt * 16 + lm) * CC + h * HEAD;
  v16h af;
  load8h<0>(af, qp + 16 * l16);
  load8h<8>(af, qp + 16 * l16 + 8);

  // per-row (C-layout rows of this lane) bias pointer and region id
  const float* brow[8];
  int rreg[8];
#pragma unroll
  for (int vj = 0; vj < 8; ++vj) {
    int m = qt * 16 + vj + 8 * l16;
    rreg[vj] = regid[w * NP + m];
    brow[vj] = biasT + ((size_t)h * NP + m) * NP;
  }

  v8f o0 = {}, o1 = {};
  float mrow[8], lrow[8];
#pragma unroll
  for (int vj = 0; vj < 8; ++vj) { mrow[vj] = -3.0e38f; lrow[vj] = 0.0f; }

  const half_t* vbase = vT + ((size_t)w * NHD + h) * HEAD * NP;

  for (int kt = 0; kt < 13; ++kt) {
    int kb = kt * 32;
    // prefetch next iteration's K rows and V columns into cache (global_prefetch_b8)
    if (kt < 12) {
      __builtin_prefetch(k + ((size_t)w * NP + kb + 32 + lm) * CC + h * HEAD, 0, 1);
      __builtin_prefetch(k + ((size_t)w * NP + kb + 48 + lm) * CC + h * HEAD, 0, 1);
      __builtin_prefetch(vbase + (size_t)lm * NP + kb + 32, 0, 1);
      __builtin_prefetch(vbase + (size_t)(16 + lm) * NP + kb + 32, 0, 1);
    }
    // --- scores S = Q K^T for 32 key tokens (two 16-col tiles) ---
    v8f s0 = {}, s1 = {};
    {
      const half_t* kp = k + ((size_t)w * NP + kb + lm) * CC + h * HEAD + 16 * l16;
      v16h bf; load8h<0>(bf, kp); load8h<8>(bf, kp + 8);
      s0 = wmma_f16(af, bf, s0);
    }
    {
      const half_t* kp = k + ((size_t)w * NP + kb + 16 + lm) * CC + h * HEAD + 16 * l16;
      v16h bf; load8h<0>(bf, kp); load8h<8>(bf, kp + 8);
      s1 = wmma_f16(af, bf, s1);
    }
    int rc0 = regid[w * NP + kb + lm];
    int rc1 = regid[w * NP + kb + 16 + lm];
    // --- bias + mask + online softmax update ---
#pragma unroll
    for (int vj = 0; vj < 8; ++vj) {
      float sa = s0[vj] + brow[vj][kb + lm]      + (rreg[vj] == rc0 ? 0.0f : -100.0f);
      float sb = s1[vj] + brow[vj][kb + 16 + lm] + (rreg[vj] == rc1 ? 0.0f : -100.0f);
      float t = fmaxf(sa, sb);
      t = fmaxf(t, __shfl_xor(t, 1));
      t = fmaxf(t, __shfl_xor(t, 2));
      t = fmaxf(t, __shfl_xor(t, 4));
      t = fmaxf(t, __shfl_xor(t, 8));
      float nm = fmaxf(mrow[vj], t);
      float corr = __expf(mrow[vj] - nm);
      float pa = __expf(sa - nm);
      float pb = __expf(sb - nm);
      float rsum = pa + pb;
      rsum += __shfl_xor(rsum, 1);
      rsum += __shfl_xor(rsum, 2);
      rsum += __shfl_xor(rsum, 4);
      rsum += __shfl_xor(rsum, 8);
      lrow[vj] = lrow[vj] * corr + rsum;
      mrow[vj] = nm;
      o0[vj] = o0[vj] * corr;
      o1[vj] = o1[vj] * corr;
      pSh[vj + 8 * l16][lm]      = (half_t)pa;
      pSh[vj + 8 * l16][16 + lm] = (half_t)pb;
    }
    __syncthreads();
    // P as A-fragment (16 rows x 32 key tokens)
    v16h pf;
    load8h<0>(pf, &pSh[lm][16 * l16]);
    load8h<8>(pf, &pSh[lm][16 * l16 + 8]);
    // V B-fragments from transposed V (k-dim contiguous)
    const half_t* vp0 = vbase + (size_t)lm * NP        + kb + 16 * l16;
    const half_t* vp1 = vbase + (size_t)(16 + lm) * NP + kb + 16 * l16;
    v16h vf0, vf1;
    load8h<0>(vf0, vp0); load8h<8>(vf0, vp0 + 8);
    load8h<0>(vf1, vp1); load8h<8>(vf1, vp1 + 8);
    o0 = wmma_f16(pf, vf0, o0);
    o1 = wmma_f16(pf, vf1, o1);
    __syncthreads();
  }
  // normalize + store y (f16, row-major (NWIN*NP, C))
#pragma unroll
  for (int vj = 0; vj < 8; ++vj) {
    float inv = 1.0f / lrow[vj];
    int m = qt * 16 + vj + 8 * l16;
    half_t* yp = y + ((size_t)w * NP + m) * CC + h * HEAD;
    yp[lm]      = (half_t)(o0[vj] * inv);
    yp[16 + lm] = (half_t)(o1[vj] * inv);
  }
}

// ---------------- host launcher ----------------
extern "C" void kernel_launch(void* const* d_in, const int* in_sizes, int n_in,
                              void* d_out, int out_size, void* d_ws, size_t ws_size,
                              hipStream_t stream) {
  const float* x        = (const float*)d_in[0];
  const float* n1w      = (const float*)d_in[1];
  const float* n1b      = (const float*)d_in[2];
  const float* qkv_w    = (const float*)d_in[3];
  const float* qkv_b    = (const float*)d_in[4];
  const float* rel_tab  = (const float*)d_in[5];
  const float* proj_w   = (const float*)d_in[6];
  const float* proj_b   = (const float*)d_in[7];
  const float* n2w      = (const float*)d_in[8];
  const float* n2b      = (const float*)d_in[9];
  const float* fc1_w    = (const float*)d_in[10];
  const float* fc1_b    = (const float*)d_in[11];
  const float* fc2_w    = (const float*)d_in[12];
  const float* fc2_b    = (const float*)d_in[13];
  float* out = (float*)d_out;

  char* ws = (char*)d_ws;
  size_t off = 0;
  auto alloc = [&](size_t bytes) -> void* {
    void* p = ws + off;
    off += (bytes + 255) & ~(size_t)255;
    return p;
  };
  half_t* xw   = (half_t*)alloc((size_t)MPAD * CC * 2);
  half_t* qb   = (half_t*)alloc((size_t)MPAD * CC * 2);
  half_t* kb   = (half_t*)alloc((size_t)MPAD * CC * 2);
  half_t* vT   = (half_t*)alloc((size_t)NWIN * NHD * HEAD * NP * 2);
  half_t* yb   = (half_t*)alloc((size_t)MPAD * CC * 2);
  float*  biasT= (float*) alloc((size_t)NHD * NP * NP * 4);
  int*    regid= (int*)   alloc((size_t)NWIN * NP * 4);
  half_t* m1   = (half_t*)alloc((size_t)TOK * CC * 2);
  half_t* h1   = (half_t*)alloc((size_t)TOK * HID * 2);
  half_t* wqkv = (half_t*)alloc((size_t)384 * 128 * 2);
  half_t* wproj= (half_t*)alloc((size_t)128 * 128 * 2);
  half_t* wfc1 = (half_t*)alloc((size_t)512 * 128 * 2);
  half_t* wfc2 = (half_t*)alloc((size_t)128 * 512 * 2);

  // 1) f16 weight copies
  cvt_f32_f16<<<(49152 + 255) / 256, 256, 0, stream>>>(qkv_w, wqkv, 49152);
  cvt_f32_f16<<<(16384 + 255) / 256, 256, 0, stream>>>(proj_w, wproj, 16384);
  cvt_f32_f16<<<(65536 + 255) / 256, 256, 0, stream>>>(fc1_w, wfc1, 65536);
  cvt_f32_f16<<<(65536 + 255) / 256, 256, 0, stream>>>(fc2_w, wfc2, 65536);

  // 2) bias table + region ids
  build_bias<<<(NHD * NP * NP + 255) / 256, 256, 0, stream>>>(rel_tab, biasT);
  build_regid<<<(NWIN * NP + 255) / 256, 256, 0, stream>>>(regid);

  // 3) LN1 + shift + window partition
  ln1_shift_win<<<MPAD, 128, 0, stream>>>(x, n1w, n1b, xw);

  // 4) QKV GEMM (M=53248, N=384, K=128)
  gemm_nt<<<dim3(MPAD / 256, 384 / 64), 256, 0, stream>>>(
      xw, wqkv, qkv_b, 128, 0, nullptr, nullptr, qb, kb, vT);

  // 5) attention
  attn_flash<<<dim3(NP / 16, NHD, NWIN), 32, 0, stream>>>(qb, kb, vT, biasT, regid, yb);

  // 6) proj GEMM + window reverse + roll + residual -> x1 in d_out
  gemm_nt<<<dim3(MPAD / 256, 128 / 64), 256, 0, stream>>>(
      yb, wproj, proj_b, 128, 1, out, x, nullptr, nullptr, nullptr);

  // 7) LN2 on x1
  ln2_plain<<<TOK, 128, 0, stream>>>(out, n2w, n2b, m1);

  // 8) fc1 + GELU (M=50176, N=512, K=128)
  gemm_nt<<<dim3(TOK / 256, HID / 64), 256, 0, stream>>>(
      m1, wfc1, fc1_b, 128, 2, nullptr, nullptr, h1, nullptr, nullptr);

  // 9) fc2 + residual into d_out (M=50176, N=128, K=512)
  gemm_nt<<<dim3(TOK / 256, 128 / 64), 256, 0, stream>>>(
      h1, wfc2, fc2_b, 512, 3, out, nullptr, nullptr, nullptr, nullptr);

  (void)in_sizes; (void)n_in; (void)out_size; (void)ws_size;
}
